// MambaBlock_64295660421774
// MI455X (gfx1250) — matrix operand order repaired and checked by reading, not compiled, not run on hardware
//
#include <hip/hip_runtime.h>
#include <hip/hip_bf16.h>
#include <math.h>

// Mamba block for MI455X (gfx1250, wave32).
// GEMMs: v_wmma_f32_16x16x32_f16, f16 operands, 1x4 (or 1x2) N-tile strips per
// wave; all operand fragments for a K-step are issued before the matrix ops so
// s_wait_loadcnt can be partial and loads overlap WMMA execution.
// Selective scan: 3-phase chunked scan; out-proj gating fused into scan epilogue.

typedef __attribute__((ext_vector_type(16))) _Float16 v16h;
typedef __attribute__((ext_vector_type(8)))  _Float16 v8h;
typedef __attribute__((ext_vector_type(8)))  float    v8f;
typedef __attribute__((ext_vector_type(4)))  float    v4f;

constexpr int cB  = 2;
constexpr int cT  = 1024;
constexpr int cDM = 512;
constexpr int cDS = 16;
constexpr int cK  = 4;
constexpr int cDI = 1024;
constexpr int cBT = cB * cT;          // 2048 rows
constexpr int NCH = 16;               // scan chunks
constexpr int CHL = cT / NCH;         // 64 steps per chunk

__device__ __forceinline__ float sigmoidf_(float v) { return 1.f / (1.f + expf(-v)); }
__device__ __forceinline__ float siluf_(float v)    { return v * sigmoidf_(v); }
__device__ __forceinline__ float softplusf_(float v){ return (v > 20.f) ? v : log1pf(expf(v)); }

// ---------------------------------------------------------------------------
// 1xW strip of 16x16 output tiles of A(MxK) @ W^T, f16 row-major operands.
// A layout (16-bit 16x32): lane<16 -> K {0..7}U{16..23}; lane>=16 -> {8..15}U{24..31}
// B layout (32x16):        lane<16 -> K {0..7}U{16..23}; lane>=16 -> {8..15}U{24..31}
// Per K-step: issue 2 A b128 loads + 2*TW B b128 loads, then TW WMMAs.
// ---------------------------------------------------------------------------
template <int TW>
__device__ __forceinline__ void wmma_strip(const _Float16* __restrict__ A, int lda,
                                           const _Float16* __restrict__ Wm, int ldw,
                                           int m0, int n0, int Kd, v8f acc[TW]) {
  const int lane = threadIdx.x & 31;
  const int lh   = lane >> 4;
  const _Float16* arow = A + (size_t)(m0 + (lane & 15)) * lda;
  const _Float16* wrow[TW];
#pragma unroll
  for (int w = 0; w < TW; ++w) {
    wrow[w] = Wm + (size_t)(n0 + 16 * w + (lane & 15)) * ldw;
    acc[w] = (v8f){0.f, 0.f, 0.f, 0.f, 0.f, 0.f, 0.f, 0.f};
  }
  for (int k = 0; k < Kd; k += 32) {
    const int ka0 = k + (lh ? 8 : 0);
    const int ka1 = k + (lh ? 24 : 16);
    const int kb  = k + (lh ? 8 : 0);
    // ---- load phase: A fragment + all TW B fragments ----
    const v8h alo = *(const v8h*)(arow + ka0);
    const v8h ahi = *(const v8h*)(arow + ka1);
    v8h blo[TW], bhi[TW];
#pragma unroll
    for (int w = 0; w < TW; ++w) {
      blo[w] = *(const v8h*)(wrow[w] + kb);
      bhi[w] = *(const v8h*)(wrow[w] + kb + 16);
    }
    // ---- compute phase: TW WMMAs reusing the A fragment ----
    const v16h a = __builtin_shufflevector(alo, ahi, 0, 1, 2, 3, 4, 5, 6, 7,
                                           8, 9, 10, 11, 12, 13, 14, 15);
#pragma unroll
    for (int w = 0; w < TW; ++w) {
      const v16h b = __builtin_shufflevector(blo[w], bhi[w], 0, 1, 2, 3, 4, 5, 6, 7,
                                             8, 9, 10, 11, 12, 13, 14, 15);
      acc[w] = __builtin_amdgcn_wmma_f32_16x16x32_f16(false, a, false, b, (short)0,
                                                      acc[w], false, false);
    }
  }
}

// ---------------------------------------------------------------------------
// Kernel 0: f32 -> f16 conversion (weights / input, one shot).
// ---------------------------------------------------------------------------
__global__ void k_cvt16(const float* __restrict__ src, _Float16* __restrict__ dst, int n) {
  const int g = blockIdx.x * blockDim.x + threadIdx.x;
  if (g < n) dst[g] = (_Float16)src[g];
}

// ---------------------------------------------------------------------------
// Kernel 1: xz = x @ in_w^T + in_b ; split into x_ and z halves (f32).
// M=2048, N=2048, K=512.  Wave strip = 16x64; 4 waves/block.
// ---------------------------------------------------------------------------
__global__ void k_inproj(const _Float16* __restrict__ xh, const _Float16* __restrict__ in_wh,
                         const float* __restrict__ in_b,
                         float* __restrict__ xbuf, float* __restrict__ zbuf) {
  const int NG = (2 * cDI) / 64;                  // 32 strip groups
  const int strip = blockIdx.x * (blockDim.x >> 5) + (threadIdx.x >> 5);
  if (strip >= (cBT / 16) * NG) return;           // wave-uniform
  const int m0 = (strip / NG) * 16, n0 = (strip % NG) * 64;
  v8f acc[4];
  wmma_strip<4>(xh, cDM, in_wh, cDM, m0, n0, cDM, acc);
  const int lane = threadIdx.x & 31;
  const int lh   = lane >> 4;
  // n0 is 64-aligned and DI is a multiple of 64: whole strip in one half.
  float* __restrict__ dst = (n0 < cDI) ? xbuf : zbuf;
  const int cbase = (n0 < cDI) ? n0 : (n0 - cDI);
#pragma unroll
  for (int w = 0; w < 4; ++w) {
    const int col = cbase + 16 * w + (lane & 15);
    const float bias = in_b[n0 + 16 * w + (lane & 15)];
#pragma unroll
    for (int r = 0; r < 8; ++r)
      dst[(size_t)(m0 + r + 8 * lh) * cDI + col] = acc[w][r] + bias;
  }
}

// ---------------------------------------------------------------------------
// Kernel 2: depthwise causal conv (K=4) over time + SiLU.
// Writes f32 (for scan) and f16 (for x-proj GEMM A).
// ---------------------------------------------------------------------------
__global__ void k_conv_silu(const float* __restrict__ xbuf,
                            const float* __restrict__ conv_w,
                            const float* __restrict__ conv_b,
                            float* __restrict__ xc, _Float16* __restrict__ xch) {
  const int g = blockIdx.x * blockDim.x + threadIdx.x;
  if (g >= cBT * cDI) return;
  const int c  = g % cDI;
  const int bt = g / cDI;
  const int t  = bt % cT;
  float acc = conv_b[c];
#pragma unroll
  for (int j = 0; j < cK; ++j) {
    const int tt = t + j - (cK - 1);
    if (tt >= 0) acc += conv_w[c * cK + j] * xbuf[(size_t)(bt + j - (cK - 1)) * cDI + c];
  }
  const float v = siluf_(acc);
  xc[g]  = v;
  xch[g] = (_Float16)v;
}

// ---------------------------------------------------------------------------
// Kernel 3: bcd = xc @ xp_w^T + xp_b ; scatter into B_s, C_s (f32), dtp (f16).
// M=2048, N=1056, K=1024.  Wave strip = 16x32 (width 2, 33 groups); group 0
// maps exactly onto the Bs|Cs pair.
// ---------------------------------------------------------------------------
__global__ void k_xproj(const _Float16* __restrict__ xch, const _Float16* __restrict__ xp_wh,
                        const float* __restrict__ xp_b,
                        float* __restrict__ Bs, float* __restrict__ Cs,
                        _Float16* __restrict__ dtph) {
  const int NG = (2 * cDS + cDI) / 32;            // 33
  const int strip = blockIdx.x * (blockDim.x >> 5) + (threadIdx.x >> 5);
  if (strip >= (cBT / 16) * NG) return;
  const int m0 = (strip / NG) * 16, n0 = (strip % NG) * 32;
  v8f acc[2];
  wmma_strip<2>(xch, cDI, xp_wh, cDI, m0, n0, cDI, acc);
  const int lane = threadIdx.x & 31;
  const int lh   = lane >> 4;
  const int ln15 = lane & 15;
#pragma unroll
  for (int w = 0; w < 2; ++w) {
    const int col = n0 + 16 * w + ln15;           // n0, w wave-uniform
    const float bias = xp_b[col];
    if (n0 == 0 && w == 0) {
#pragma unroll
      for (int r = 0; r < 8; ++r)
        Bs[(size_t)(m0 + r + 8 * lh) * cDS + ln15] = acc[w][r] + bias;
    } else if (n0 == 0 && w == 1) {
#pragma unroll
      for (int r = 0; r < 8; ++r)
        Cs[(size_t)(m0 + r + 8 * lh) * cDS + ln15] = acc[w][r] + bias;
    } else {
      const int cc = col - 2 * cDS;
#pragma unroll
      for (int r = 0; r < 8; ++r)
        dtph[(size_t)(m0 + r + 8 * lh) * cDI + cc] = (_Float16)(acc[w][r] + bias);
    }
  }
}

// ---------------------------------------------------------------------------
// Kernel 4: dt = softplus(dtp @ dt_w^T + dt_b) -> f32.  M=2048, N=1024, K=1024.
// ---------------------------------------------------------------------------
__global__ void k_dtproj(const _Float16* __restrict__ dtph, const _Float16* __restrict__ dt_wh,
                         const float* __restrict__ dt_b, float* __restrict__ dtv) {
  const int NG = cDI / 64;                        // 16
  const int strip = blockIdx.x * (blockDim.x >> 5) + (threadIdx.x >> 5);
  if (strip >= (cBT / 16) * NG) return;
  const int m0 = (strip / NG) * 16, n0 = (strip % NG) * 64;
  v8f acc[4];
  wmma_strip<4>(dtph, cDI, dt_wh, cDI, m0, n0, cDI, acc);
  const int lane = threadIdx.x & 31;
  const int lh   = lane >> 4;
#pragma unroll
  for (int w = 0; w < 4; ++w) {
    const int col = n0 + 16 * w + (lane & 15);
    const float bias = dt_b[col];
#pragma unroll
    for (int r = 0; r < 8; ++r)
      dtv[(size_t)(m0 + r + 8 * lh) * cDI + col] = softplusf_(acc[w][r] + bias);
  }
}

// ---------------------------------------------------------------------------
// Kernel 5: scan phase 1 — per-chunk (prod dA, local h end-state), h0 = 0.
// thread <- (b, chunk, c); 32768 threads.
// ---------------------------------------------------------------------------
__global__ void k_scan_chunk(const float* __restrict__ dtv, const float* __restrict__ xc,
                             const float* __restrict__ Bs, const float* __restrict__ A_log,
                             float* __restrict__ cumA, float* __restrict__ hEnd) {
  const int g = blockIdx.x * blockDim.x + threadIdx.x;
  if (g >= cB * NCH * cDI) return;
  const int c  = g % cDI;
  const int ch = (g / cDI) % NCH;
  const int b  = g / (cDI * NCH);
  float Aval[cDS], pA[cDS], h[cDS];
#pragma unroll
  for (int s = 0; s < cDS; ++s) {
    Aval[s] = -expf(A_log[c * cDS + s]);
    pA[s] = 1.f;
    h[s]  = 0.f;
  }
  const int t0 = ch * CHL;
  for (int t = t0; t < t0 + CHL; ++t) {
    const size_t rc = (size_t)(b * cT + t);
    const float dt = dtv[rc * cDI + c];
    const float dx = dt * xc[rc * cDI + c];
    const v4f* bp = (const v4f*)(Bs + rc * cDS);  // 64B-aligned row
    v4f bb[4];
#pragma unroll
    for (int q = 0; q < 4; ++q) bb[q] = bp[q];
#pragma unroll
    for (int s = 0; s < cDS; ++s) {
      const float dA = expf(dt * Aval[s]);
      h[s]  = dA * h[s] + dx * bb[s >> 2][s & 3];
      pA[s] *= dA;
    }
  }
  const size_t base = ((size_t)(b * NCH + ch) * cDI + c) * cDS;
#pragma unroll
  for (int s = 0; s < cDS; ++s) { cumA[base + s] = pA[s]; hEnd[base + s] = h[s]; }
}

// ---------------------------------------------------------------------------
// Kernel 6: scan phase 2 — serial combine across the 16 chunks.
// ---------------------------------------------------------------------------
__global__ void k_scan_combine(const float* __restrict__ cumA,
                               const float* __restrict__ hEnd,
                               float* __restrict__ Hini) {
  const int g = blockIdx.x * blockDim.x + threadIdx.x;
  if (g >= cB * cDI * cDS) return;
  const int s = g % cDS;
  const int c = (g / cDS) % cDI;
  const int b = g / (cDS * cDI);
  float H = 0.f;
  for (int k = 0; k < NCH; ++k) {
    const size_t idx = ((size_t)(b * NCH + k) * cDI + c) * cDS + s;
    Hini[idx] = H;
    H = cumA[idx] * H + hEnd[idx];
  }
}

// ---------------------------------------------------------------------------
// Kernel 7: scan phase 3 — replay chunk with correct initial state; fuse
// out-proj gating: write g = y * silu(z) directly as f16 GEMM operand.
// ---------------------------------------------------------------------------
__global__ void k_scan_final(const float* __restrict__ dtv, const float* __restrict__ xc,
                             const float* __restrict__ Bs, const float* __restrict__ Cs,
                             const float* __restrict__ A_log, const float* __restrict__ D_skip,
                             const float* __restrict__ Hini, const float* __restrict__ zbuf,
                             _Float16* __restrict__ gbuf) {
  const int g = blockIdx.x * blockDim.x + threadIdx.x;
  if (g >= cB * NCH * cDI) return;
  const int c  = g % cDI;
  const int ch = (g / cDI) % NCH;
  const int b  = g / (cDI * NCH);
  float Aval[cDS], h[cDS];
  const size_t base = ((size_t)(b * NCH + ch) * cDI + c) * cDS;
#pragma unroll
  for (int s = 0; s < cDS; ++s) {
    Aval[s] = -expf(A_log[c * cDS + s]);
    h[s]    = Hini[base + s];
  }
  const float Dv = D_skip[c];
  const int t0 = ch * CHL;
  for (int t = t0; t < t0 + CHL; ++t) {
    const size_t rc = (size_t)(b * cT + t);
    const float dt = dtv[rc * cDI + c];
    const float xv = xc[rc * cDI + c];
    const float dx = dt * xv;
    float y = Dv * xv;
    const v4f* bp = (const v4f*)(Bs + rc * cDS);
    const v4f* cp = (const v4f*)(Cs + rc * cDS);
    v4f bb[4], cc[4];
#pragma unroll
    for (int q = 0; q < 4; ++q) { bb[q] = bp[q]; cc[q] = cp[q]; }
#pragma unroll
    for (int s = 0; s < cDS; ++s) {
      const float dA = expf(dt * Aval[s]);
      h[s] = dA * h[s] + dx * bb[s >> 2][s & 3];
      y += h[s] * cc[s >> 2][s & 3];
    }
    const float zv = zbuf[rc * cDI + c];
    gbuf[rc * cDI + c] = (_Float16)(y * siluf_(zv));
  }
}

// ---------------------------------------------------------------------------
// Kernel 8: o = gbuf @ out_w^T + out_b + x (residual).  M=2048, N=512, K=1024.
// ---------------------------------------------------------------------------
__global__ void k_outproj(const _Float16* __restrict__ gbuf, const _Float16* __restrict__ out_wh,
                          const float* __restrict__ out_b, const float* __restrict__ x,
                          float* __restrict__ obuf) {
  const int NG = cDM / 64;                        // 8
  const int strip = blockIdx.x * (blockDim.x >> 5) + (threadIdx.x >> 5);
  if (strip >= (cBT / 16) * NG) return;
  const int m0 = (strip / NG) * 16, n0 = (strip % NG) * 64;
  v8f acc[4];
  wmma_strip<4>(gbuf, cDI, out_wh, cDI, m0, n0, cDI, acc);
  const int lane = threadIdx.x & 31;
  const int lh   = lane >> 4;
#pragma unroll
  for (int w = 0; w < 4; ++w) {
    const int col = n0 + 16 * w + (lane & 15);
    const float bias = out_b[col];
#pragma unroll
    for (int r = 0; r < 8; ++r) {
      const int m = m0 + r + 8 * lh;
      obuf[(size_t)m * cDM + col] = acc[w][r] + bias + x[(size_t)m * cDM + col];
    }
  }
}

// ---------------------------------------------------------------------------
// Kernel 9: row-wise LayerNorm over DM=512.
// ---------------------------------------------------------------------------
__global__ void k_ln(const float* __restrict__ o, const float* __restrict__ g,
                     const float* __restrict__ bta, float* __restrict__ out) {
  __shared__ float ssum[256];
  __shared__ float ssq[256];
  const int tid = threadIdx.x;
  const int row = blockIdx.x;
  const float* op = o + (size_t)row * cDM;
  const float v0 = op[tid];
  const float v1 = op[tid + 256];
  ssum[tid] = v0 + v1;
  ssq[tid]  = v0 * v0 + v1 * v1;
  __syncthreads();
  for (int off = 128; off > 0; off >>= 1) {
    if (tid < off) { ssum[tid] += ssum[tid + off]; ssq[tid] += ssq[tid + off]; }
    __syncthreads();
  }
  const float mu   = ssum[0] * (1.f / cDM);
  const float var  = ssq[0] * (1.f / cDM) - mu * mu;
  const float rstd = rsqrtf(var + 1e-5f);
  out[(size_t)row * cDM + tid]       = (v0 - mu) * rstd * g[tid] + bta[tid];
  out[(size_t)row * cDM + tid + 256] = (v1 - mu) * rstd * g[tid + 256] + bta[tid + 256];
}

// ---------------------------------------------------------------------------
extern "C" void kernel_launch(void* const* d_in, const int* in_sizes, int n_in,
                              void* d_out, int out_size, void* d_ws, size_t ws_size,
                              hipStream_t stream) {
  const float* x      = (const float*)d_in[0];
  const float* in_w   = (const float*)d_in[1];
  const float* in_b   = (const float*)d_in[2];
  const float* conv_w = (const float*)d_in[3];
  const float* conv_b = (const float*)d_in[4];
  const float* xp_w   = (const float*)d_in[5];
  const float* xp_b   = (const float*)d_in[6];
  const float* dt_w   = (const float*)d_in[7];
  const float* dt_b   = (const float*)d_in[8];
  const float* A_log  = (const float*)d_in[9];
  const float* D_skip = (const float*)d_in[10];
  const float* out_w  = (const float*)d_in[11];
  const float* out_b  = (const float*)d_in[12];
  const float* ln_g   = (const float*)d_in[13];
  const float* ln_b   = (const float*)d_in[14];
  float* out = (float*)d_out;

  // ---- workspace carve-up ----
  const size_t ACT = (size_t)cBT * cDI;            // 2M elements
  const size_t SUM = (size_t)cB * NCH * cDI * cDS; // 512K
  float* xbuf = (float*)d_ws;                      // f32 region
  float* zbuf = xbuf + ACT;
  float* xc   = zbuf + ACT;
  float* dtv  = xc   + ACT;
  float* Bs   = dtv  + ACT;
  float* Cs   = Bs   + (size_t)cBT * cDS;
  float* cumA = Cs   + (size_t)cBT * cDS;
  float* hEnd = cumA + SUM;
  float* Hini = hEnd + SUM;
  float* obuf = Hini + SUM;
  _Float16* xh     = (_Float16*)(obuf + (size_t)cBT * cDM); // f16 region
  _Float16* in_wh  = xh     + (size_t)cBT * cDM;            // 2048*512
  _Float16* xp_wh  = in_wh  + (size_t)(2 * cDI) * cDM;      // 1056*1024
  _Float16* dt_wh  = xp_wh  + (size_t)(2 * cDS + cDI) * cDI;
  _Float16* out_wh = dt_wh  + (size_t)cDI * cDI;            // 512*1024
  _Float16* xch    = out_wh + (size_t)cDM * cDI;
  _Float16* dtph   = xch    + ACT;
  _Float16* gbuf   = dtph   + ACT;

  // ---- 0) one-shot f16 conversions (weights + input) ----
  const int nX  = cBT * cDM;               // 1048576
  const int nIW = 2 * cDI * cDM;           // 1048576
  const int nXP = (2 * cDS + cDI) * cDI;   // 1081344
  const int nDT = cDI * cDI;               // 1048576
  const int nOW = cDM * cDI;               // 524288
  k_cvt16<<<(nX  + 255) / 256, 256, 0, stream>>>(x,     xh,     nX);
  k_cvt16<<<(nIW + 255) / 256, 256, 0, stream>>>(in_w,  in_wh,  nIW);
  k_cvt16<<<(nXP + 255) / 256, 256, 0, stream>>>(xp_w,  xp_wh,  nXP);
  k_cvt16<<<(nDT + 255) / 256, 256, 0, stream>>>(dt_w,  dt_wh,  nDT);
  k_cvt16<<<(nOW + 255) / 256, 256, 0, stream>>>(out_w, out_wh, nOW);

  // ---- pipeline ----
  // in-proj: 128 M-tiles x 32 strip groups = 4096 waves, 4 waves/block
  k_inproj<<<(128 * 32) / 4, 128, 0, stream>>>(xh, in_wh, in_b, xbuf, zbuf);
  k_conv_silu<<<(cBT * cDI) / 256, 256, 0, stream>>>(xbuf, conv_w, conv_b, xc, xch);
  // x-proj: 128 x 33 strips (width 2)
  k_xproj<<<(128 * 33) / 4, 128, 0, stream>>>(xch, xp_wh, xp_b, Bs, Cs, dtph);
  // dt-proj: 128 x 16 strips (width 4)
  k_dtproj<<<(128 * 16) / 4, 128, 0, stream>>>(dtph, dt_wh, dt_b, dtv);
  k_scan_chunk<<<(cB * NCH * cDI) / 256, 256, 0, stream>>>(dtv, xc, Bs, A_log, cumA, hEnd);
  k_scan_combine<<<(cB * cDI * cDS) / 256, 256, 0, stream>>>(cumA, hEnd, Hini);
  k_scan_final<<<(cB * NCH * cDI) / 256, 256, 0, stream>>>(dtv, xc, Bs, Cs, A_log, D_skip,
                                                           Hini, zbuf, gbuf);
  // out-proj: 128 x 8 strips (width 4)
  k_outproj<<<(128 * 8) / 4, 128, 0, stream>>>(gbuf, out_wh, out_b, x, obuf);
  k_ln<<<cBT, 256, 0, stream>>>(obuf, ln_g, ln_b, out);
}
